// ASSA_68075231642102
// MI455X (gfx1250) — compile-verified, matrix operand order checked by reading, not run on hardware
//
#include <hip/hip_runtime.h>
#include <hip/hip_bf16.h>

// ---------------- CDNA5 WMMA types ----------------
typedef __attribute__((ext_vector_type(16))) __bf16 v16bf;
typedef __attribute__((ext_vector_type(8)))  float  v8f;

#define HID   512
#define SLEN  4096
#define BATCH 2
#define NHEAD 8
#define DH    64
#define MROWS (BATCH * SLEN)   // 8192

// A/B fragment K-index for 16-bit 16x32 (A) / 32x16 (B) per ISA 7.12.2.
// Per lane this yields two contiguous 8-element runs -> b128-vectorizable.
__device__ __forceinline__ int kmap(int e, int lane) {
    int i = e >> 1;
    return ((i & 4) << 2) | ((lane & 16) >> 1) | ((i & 3) << 1) | (e & 1);
}

__device__ __forceinline__ v8f wmma_bf16(v16bf a, v16bf b, v8f c) {
    return __builtin_amdgcn_wmma_f32_16x16x32_bf16(
        false, a, false, b, (short)0, c, false, false);
}

// async global->LDS copy, 16 bytes per lane (gfx1250, tracked by ASYNCcnt;
// async loads complete in order -> usable for double buffering)
__device__ __forceinline__ void async_b128(unsigned lds_off, unsigned long long gaddr) {
    asm volatile("global_load_async_to_lds_b128 %0, %1, off"
                 :: "v"(lds_off), "v"(gaddr) : "memory");
}
__device__ __forceinline__ void async_b256(unsigned lds_off, unsigned long long gaddr) {
    asm volatile("global_load_async_to_lds_b128 %0, %1, off\n\t"
                 "global_load_async_to_lds_b128 %0, %1, off offset:16"
                 :: "v"(lds_off), "v"(gaddr) : "memory");
}
__device__ __forceinline__ void wait_async_le0() {
    asm volatile("s_wait_asynccnt 0x0" ::: "memory");
}
__device__ __forceinline__ void wait_async_le1() {
    asm volatile("s_wait_asynccnt 0x1" ::: "memory");
}
__device__ __forceinline__ void wait_async_le4() {
    asm volatile("s_wait_asynccnt 0x4" ::: "memory");
}

// ---------------- f32 -> bf16 convert ----------------
__global__ void cvt_bf16_kernel(const float* __restrict__ in,
                                __bf16* __restrict__ out, int n) {
    int i = blockIdx.x * blockDim.x + threadIdx.x;
    if (i < n) out[i] = (__bf16)in[i];
}

// ---------------- register-blocked 32x64 GEMM core ----------------
// acc[i][j]: 16x16 tile at rows m0+i*16, cols n0+j*16.  y = A * W^T
__device__ __forceinline__ void gemm_core_32x64(
    const __bf16* __restrict__ A, const __bf16* __restrict__ W,
    int m0, int n0, int lane, v8f acc[2][4]) {
    const int ar0 = m0 + (lane & 15);
    const int ar1 = ar0 + 16;
    const int nc  = n0 + (lane & 15);
    for (int k0 = 0; k0 < HID; k0 += 32) {
        v16bf a0, a1, b0, b1, b2, b3;
#pragma unroll
        for (int e = 0; e < 16; ++e) {
            int k = k0 + kmap(e, lane);
            a0[e] = A[(size_t)ar0 * HID + k];
            a1[e] = A[(size_t)ar1 * HID + k];
            b0[e] = W[(size_t)(nc +  0) * HID + k];   // B(k,n) = W[n][k]
            b1[e] = W[(size_t)(nc + 16) * HID + k];
            b2[e] = W[(size_t)(nc + 32) * HID + k];
            b3[e] = W[(size_t)(nc + 48) * HID + k];
        }
        acc[0][0] = wmma_bf16(a0, b0, acc[0][0]);
        acc[0][1] = wmma_bf16(a0, b1, acc[0][1]);
        acc[0][2] = wmma_bf16(a0, b2, acc[0][2]);
        acc[0][3] = wmma_bf16(a0, b3, acc[0][3]);
        acc[1][0] = wmma_bf16(a1, b0, acc[1][0]);
        acc[1][1] = wmma_bf16(a1, b1, acc[1][1]);
        acc[1][2] = wmma_bf16(a1, b2, acc[1][2]);
        acc[1][3] = wmma_bf16(a1, b3, acc[1][3]);
    }
}

// ---------------- QKV projection: out[B,H,S,DH] bf16, scaled ----------------
__global__ __launch_bounds__(128)
void proj_kernel(const __bf16* __restrict__ X, const __bf16* __restrict__ W,
                 const float* __restrict__ bias, __bf16* __restrict__ out,
                 float scale) {
    const int lane = threadIdx.x & 31;
    const int wave = threadIdx.x >> 5;
    const int tile = blockIdx.x * 4 + wave;          // (8192/32)*(512/64)=2048
    if (tile >= (MROWS / 32) * (HID / 64)) return;
    const int m0 = (tile >> 3) << 5;
    const int n0 = (tile & 7) << 6;
    __builtin_prefetch(W + (size_t)n0 * HID, 0, 1);
    v8f acc[2][4] = {{{}, {}, {}, {}}, {{}, {}, {}, {}}};
    gemm_core_32x64(X, W, m0, n0, lane, acc);
#pragma unroll
    for (int i = 0; i < 2; ++i) {
#pragma unroll
        for (int j = 0; j < 4; ++j) {
#pragma unroll
            for (int r = 0; r < 8; ++r) {
                int row = m0 + i * 16 + r + ((lane & 16) >> 1);
                int col = n0 + j * 16 + (lane & 15);
                int b = row >> 12, s = row & (SLEN - 1);
                int h = col >> 6,  d = col & (DH - 1);
                float v = (acc[i][j][r] + bias[col]) * scale;
                out[(((size_t)(b * NHEAD + h) * SLEN + s) * DH) + d] = (__bf16)v;
            }
        }
    }
}

// ---------------- output GEMM: out = ctx * Wo^T + bo (f32) ----------------
__global__ __launch_bounds__(128)
void ogemm_kernel(const __bf16* __restrict__ A, const __bf16* __restrict__ W,
                  const float* __restrict__ bias, float* __restrict__ out) {
    const int lane = threadIdx.x & 31;
    const int wave = threadIdx.x >> 5;
    const int tile = blockIdx.x * 4 + wave;
    if (tile >= (MROWS / 32) * (HID / 64)) return;
    const int m0 = (tile >> 3) << 5;
    const int n0 = (tile & 7) << 6;
    __builtin_prefetch(W + (size_t)n0 * HID, 0, 1);
    v8f acc[2][4] = {{{}, {}, {}, {}}, {{}, {}, {}, {}}};
    gemm_core_32x64(A, W, m0, n0, lane, acc);
#pragma unroll
    for (int i = 0; i < 2; ++i) {
#pragma unroll
        for (int j = 0; j < 4; ++j) {
#pragma unroll
            for (int r = 0; r < 8; ++r) {
                int row = m0 + i * 16 + r + ((lane & 16) >> 1);
                int col = n0 + j * 16 + (lane & 15);
                out[(size_t)row * HID + col] = acc[i][j][r] + bias[col];
            }
        }
    }
}

// ---------------- attention ----------------
// Block = 4 waves = 64 consecutive query rows of one (b,h).  K/V tiles staged
// once per block into double-buffered LDS via async copies (DMA of tile t+1
// overlaps compute of tile t; ASYNCcnt completes in order).  Two passes: the
// prob>0.1 threshold needs the final softmax max m and denominator Z.
__global__ __launch_bounds__(128)
void attn_kernel(const __bf16* __restrict__ Q, const __bf16* __restrict__ K,
                 const __bf16* __restrict__ V, __bf16* __restrict__ ctx) {
    __shared__ alignas(16) __bf16 kbuf[2][32 * DH];     // 2 x 4 KB
    __shared__ alignas(16) __bf16 vbuf[2][32 * DH];     // 2 x 4 KB
    __shared__ alignas(16) __bf16 plds[4][16 * 32];     // 4 KB, per-wave P tile

    const int tid  = threadIdx.x;
    const int lane = tid & 31;
    const int wave = tid >> 5;
    const int blk  = blockIdx.x;                 // 16 bh * 64 chunks = 1024
    const int bh   = blk >> 6;
    const int q0   = ((blk & 63) << 6) + (wave << 4);
    const __bf16* Qp = Q + (size_t)bh * SLEN * DH;
    const __bf16* Kp = K + (size_t)bh * SLEN * DH;
    const __bf16* Vp = V + (size_t)bh * SLEN * DH;
    __bf16* pbuf = plds[wave];

    // Q A-fragments (16x64 = two 32-wide k-halves), loaded once per wave.
    const int qrow = q0 + (lane & 15);
    v16bf aq0, aq1;
#pragma unroll
    for (int e = 0; e < 16; ++e) {
        int d = kmap(e, lane);
        aq0[e] = Qp[(size_t)qrow * DH + d];
        aq1[e] = Qp[(size_t)qrow * DH + 32 + d];
    }

    // ---- pass 1: online row max m and Z = sum exp(s-m) ----
    float m_l[8], z_l[8];
#pragma unroll
    for (int r = 0; r < 8; ++r) { m_l[r] = -3.0e38f; z_l[r] = 0.0f; }

    // prologue: stage tile 0 (16 keys x 64 dims = 2 KB, 16 B per thread)
    async_b128((unsigned)(size_t)kbuf[0] + tid * 16,
               (unsigned long long)(size_t)Kp + tid * 16);
    for (int kt = 0; kt < SLEN / 16; ++kt) {
        if (kt + 1 < SLEN / 16) {                 // DMA next tile under compute
            async_b128((unsigned)(size_t)kbuf[(kt + 1) & 1] + tid * 16,
                       (unsigned long long)(size_t)(Kp + (size_t)(kt + 1) * 16 * DH)
                           + tid * 16);
            wait_async_le1();                     // tile kt's copy done (in order)
        } else {
            wait_async_le0();
        }
        __syncthreads();                          // all waves' copies visible
        const __bf16* kb = kbuf[kt & 1];
        v16bf bk0, bk1;
        const int krow = lane & 15;
#pragma unroll
        for (int e = 0; e < 16; ++e) {
            int d = kmap(e, lane);
            bk0[e] = kb[krow * DH + d];
            bk1[e] = kb[krow * DH + 32 + d];
        }
        v8f c = {};
        c = wmma_bf16(aq0, bk0, c);
        c = wmma_bf16(aq1, bk1, c);
#pragma unroll
        for (int r = 0; r < 8; ++r) {
            float s  = c[r];
            float mn = fmaxf(m_l[r], s);
            z_l[r] = z_l[r] * __expf(m_l[r] - mn) + __expf(s - mn);
            m_l[r] = mn;
        }
        __syncthreads();                          // safe to overwrite this buffer
    }
    // butterfly reduce (m,Z) across the 16 lanes holding each row's columns
#pragma unroll
    for (int r = 0; r < 8; ++r) {
        float m = m_l[r], z = z_l[r];
#pragma unroll
        for (int off = 1; off < 16; off <<= 1) {
            float mo = __shfl_xor(m, off, 32);
            float zo = __shfl_xor(z, off, 32);
            float mn = fmaxf(m, mo);
            z = z * __expf(m - mn) + zo * __expf(mo - mn);
            m = mn;
        }
        m_l[r] = m; z_l[r] = z;
    }
    float thr[8];
#pragma unroll
    for (int r = 0; r < 8; ++r) thr[r] = 0.1f * z_l[r];  // p>0.1*Z <=> prob>0.1

    // ---- pass 2: masked probs, ctx = P * V ----
    v8f acc[4] = {{}, {}, {}, {}};
    float kept[8] = {0, 0, 0, 0, 0, 0, 0, 0};

    // prologue: stage K+V tile 0 (32 keys: 4 KB each; 32 B/thread/buffer)
    async_b256((unsigned)(size_t)kbuf[0] + tid * 32,
               (unsigned long long)(size_t)Kp + tid * 32);
    async_b256((unsigned)(size_t)vbuf[0] + tid * 32,
               (unsigned long long)(size_t)Vp + tid * 32);
    for (int kt2 = 0; kt2 < SLEN / 32; ++kt2) {
        if (kt2 + 1 < SLEN / 32) {
            const size_t nb = (size_t)(kt2 + 1) * 32 * DH;
            async_b256((unsigned)(size_t)kbuf[(kt2 + 1) & 1] + tid * 32,
                       (unsigned long long)(size_t)(Kp + nb) + tid * 32);
            async_b256((unsigned)(size_t)vbuf[(kt2 + 1) & 1] + tid * 32,
                       (unsigned long long)(size_t)(Vp + nb) + tid * 32);
            wait_async_le4();                     // current tile's 4 copies done
        } else {
            wait_async_le0();
        }
        __syncthreads();
        const __bf16* kb = kbuf[kt2 & 1];
        const __bf16* vb = vbuf[kt2 & 1];

#pragma unroll
        for (int half = 0; half < 2; ++half) {
            const int krow = (half << 4) + (lane & 15);
            v16bf bk0, bk1;
#pragma unroll
            for (int e = 0; e < 16; ++e) {
                int d = kmap(e, lane);
                bk0[e] = kb[krow * DH + d];
                bk1[e] = kb[krow * DH + 32 + d];
            }
            v8f c = {};
            c = wmma_bf16(aq0, bk0, c);
            c = wmma_bf16(aq1, bk1, c);
#pragma unroll
            for (int r = 0; r < 8; ++r) {
                float p  = __expf(c[r] - m_l[r]);
                float pk = (p > thr[r]) ? p : 0.0f;
                kept[r] += pk;
                int row = r + ((lane & 16) >> 1);
                pbuf[row * 32 + (half << 4) + (lane & 15)] = (__bf16)pk;
            }
        }
        asm volatile("s_wait_dscnt 0" ::: "memory");  // P stores -> reads (same wave)
        v16bf pa;
        const int prow = lane & 15;
#pragma unroll
        for (int e = 0; e < 16; ++e)
            pa[e] = pbuf[prow * 32 + kmap(e, lane)];
#pragma unroll
        for (int j = 0; j < 4; ++j) {
            const int d = (j << 4) + (lane & 15);
            v16bf bv;
#pragma unroll
            for (int e = 0; e < 16; ++e) {
                int kk = kmap(e, lane);
                bv[e] = vb[kk * DH + d];              // B(k,d) = V[key][d]
            }
            acc[j] = wmma_bf16(pa, bv, acc[j]);
        }
        __syncthreads();                          // safe to overwrite this buffer
    }
    // reduce kept-mass; denom = sum(masked p) + eps*Z  (== Z*(sum_prob+eps))
#pragma unroll
    for (int r = 0; r < 8; ++r) {
        float kv = kept[r];
#pragma unroll
        for (int off = 1; off < 16; off <<= 1) kv += __shfl_xor(kv, off, 32);
        kept[r] = kv + 1e-8f * z_l[r];
    }
    const int b = bh >> 3, h = bh & 7;
#pragma unroll
    for (int j = 0; j < 4; ++j) {
#pragma unroll
        for (int r = 0; r < 8; ++r) {
            int row = q0 + r + ((lane & 16) >> 1);
            int d   = (j << 4) + (lane & 15);
            float v = acc[j][r] / kept[r];
            ctx[((size_t)b * SLEN + row) * HID + h * DH + d] = (__bf16)v;
        }
    }
}

// ---------------- host launch ----------------
extern "C" void kernel_launch(void* const* d_in, const int* in_sizes, int n_in,
                              void* d_out, int out_size, void* d_ws, size_t ws_size,
                              hipStream_t stream) {
    const float* x  = (const float*)d_in[0];
    const float* Wq = (const float*)d_in[1];
    const float* bq = (const float*)d_in[2];
    const float* Wk = (const float*)d_in[3];
    const float* bk = (const float*)d_in[4];
    const float* Wv = (const float*)d_in[5];
    const float* bv = (const float*)d_in[6];
    const float* Wo = (const float*)d_in[7];
    const float* bo = (const float*)d_in[8];
    float* out = (float*)d_out;

    char*  ws  = (char*)d_ws;
    size_t off = 0;
    auto carve = [&](size_t bytes) -> char* {
        char* p = ws + off;
        off += (bytes + 255) & ~(size_t)255;
        return p;
    };
    __bf16* xbf = (__bf16*)carve((size_t)MROWS * HID * 2);
    __bf16* wqb = (__bf16*)carve((size_t)HID * HID * 2);
    __bf16* wkb = (__bf16*)carve((size_t)HID * HID * 2);
    __bf16* wvb = (__bf16*)carve((size_t)HID * HID * 2);
    __bf16* wob = (__bf16*)carve((size_t)HID * HID * 2);
    __bf16* Qb  = (__bf16*)carve((size_t)MROWS * HID * 2);
    __bf16* Kb  = (__bf16*)carve((size_t)MROWS * HID * 2);
    __bf16* Vb  = (__bf16*)carve((size_t)MROWS * HID * 2);
    __bf16* ctx = (__bf16*)carve((size_t)MROWS * HID * 2);
    (void)ws_size; (void)in_sizes; (void)n_in; (void)out_size;

    const int nX = MROWS * HID, nW = HID * HID;
    cvt_bf16_kernel<<<(nX + 255) / 256, 256, 0, stream>>>(x,  xbf, nX);
    cvt_bf16_kernel<<<(nW + 255) / 256, 256, 0, stream>>>(Wq, wqb, nW);
    cvt_bf16_kernel<<<(nW + 255) / 256, 256, 0, stream>>>(Wk, wkb, nW);
    cvt_bf16_kernel<<<(nW + 255) / 256, 256, 0, stream>>>(Wv, wvb, nW);
    cvt_bf16_kernel<<<(nW + 255) / 256, 256, 0, stream>>>(Wo, wob, nW);

    const int gemm_blocks = (MROWS / 32) * (HID / 64) / 4;   // 512
    proj_kernel<<<gemm_blocks, 128, 0, stream>>>(xbf, wqb, bq, Qb, 0.125f);
    proj_kernel<<<gemm_blocks, 128, 0, stream>>>(xbf, wkb, bk, Kb, 1.0f);
    proj_kernel<<<gemm_blocks, 128, 0, stream>>>(xbf, wvb, bv, Vb, 1.0f);

    const int attn_blocks = BATCH * NHEAD * (SLEN / 64);     // 1024
    attn_kernel<<<attn_blocks, 128, 0, stream>>>(Qb, Kb, Vb, ctx);

    ogemm_kernel<<<gemm_blocks, 128, 0, stream>>>(ctx, wob, bo, out);
}
